// GQABroadcast_7344394076578
// MI455X (gfx1250) — matrix operand order repaired and checked by reading, not compiled, not run on hardware
//
#include <hip/hip_runtime.h>
#include <hip/hip_bf16.h>

// ---------------- types ----------------
typedef __bf16 bf16_t;
typedef __attribute__((ext_vector_type(16))) __bf16 v16bf;
typedef __attribute__((ext_vector_type(8)))  __bf16 v8bf;
typedef __attribute__((ext_vector_type(8)))  float  v8f;
typedef __attribute__((ext_vector_type(4)))  float  v4f;

// ---------------- problem constants ----------------
#define HIDDEN 2048
#define NHEADS 32
#define NKV    8
#define HD     64
#define KVDIM  (NKV*HD)      // 512
#define BB     2
#define SS     2048
#define MTOT   (BB*SS)       // 4096

// ---------------- helpers ----------------
__device__ __forceinline__ bf16_t f2bf(float f) {
  unsigned u = __builtin_bit_cast(unsigned, f);
  unsigned r = (u + 0x7FFFu + ((u >> 16) & 1u)) >> 16;   // round-to-nearest-even
  return __builtin_bit_cast(bf16_t, (unsigned short)r);
}

__device__ __forceinline__ v8f wmma_bf16(v16bf a, v16bf b, v8f c) {
  // D = A(16x32 bf16) x B(32x16 bf16) + C(16x16 f32)
  return __builtin_amdgcn_wmma_f32_16x16x32_bf16(
      /*neg_a=*/false, a, /*neg_b=*/false, b,
      /*c_mod=*/(short)0, c, /*reuse_a=*/false, /*reuse_b=*/false);
}

// load 16 bf16 as two 16B chunks (p0: elems 0..7, p1: 8..15)
__device__ __forceinline__ v16bf load_frag(const bf16_t* p0, const bf16_t* p1) {
  v16bf r;
  const v8bf a = *(const v8bf*)p0;
  const v8bf b = *(const v8bf*)p1;
#pragma unroll
  for (int j = 0; j < 8; ++j) { r[j] = a[j]; r[j + 8] = b[j]; }
  return r;
}

// ---- 16-lane butterfly reductions via DPP16 (no LDS traffic) --------------
// masks {^1 (quad_perm 1,0,3,2), ^2 (quad_perm 2,3,0,1), ^7 (row_half_mirror),
// ^15 (row_mirror)} span GF(2)^4 -> full reduction over each 16-lane group.
template <int CTRL>
__device__ __forceinline__ float dpp_step(float x) {
  int v = __builtin_bit_cast(int, x);
  int t = __builtin_amdgcn_update_dpp(v, v, CTRL, 0xF, 0xF, false);
  return __builtin_bit_cast(float, t);
}
__device__ __forceinline__ float red_max16(float x) {
  x = fmaxf(x, dpp_step<0xB1>(x));   // ^1
  x = fmaxf(x, dpp_step<0x4E>(x));   // ^2
  x = fmaxf(x, dpp_step<0x141>(x));  // ^7  row_half_mirror
  x = fmaxf(x, dpp_step<0x140>(x));  // ^15 row_mirror
  return x;
}
__device__ __forceinline__ float red_sum16(float x) {
  x += dpp_step<0xB1>(x);
  x += dpp_step<0x4E>(x);
  x += dpp_step<0x141>(x);
  x += dpp_step<0x140>(x);
  return x;
}

// ---------------- f32 -> bf16 convert (row-major) --------------------------
__global__ void k_cvt(const float* __restrict__ in, bf16_t* __restrict__ out, int n4) {
  int i = blockIdx.x * 256 + threadIdx.x;
  if (i < n4) {
    v4f v = ((const v4f*)in)[i];
    unsigned long long pk = 0;
#pragma unroll
    for (int j = 0; j < 4; ++j)
      pk |= (unsigned long long)__builtin_bit_cast(unsigned short, f2bf(v[j])) << (16 * j);
    ((unsigned long long*)out)[i] = pk;
  }
}

// ---------------- f32 [K][N] -> bf16 [N][K] transpose-convert ---------------
__global__ void k_cvt_t(const float* __restrict__ in, bf16_t* __restrict__ out,
                        int K, int N) {
  int i = blockIdx.x * 256 + threadIdx.x;
  if (i < K * N) {
    int k = i / N, n = i - k * N;
    out[(size_t)n * K + k] = f2bf(in[i]);
  }
}

// ---------------- WMMA GEMM: C[MxN] = A[MxK] * BT[NxK]^T (+bias)*alpha -----
// block tile 256(M)x128(N), K-step 32, 256 threads = 8 waves (4 M x 2 N).
// Each wave: 64x64 register tile = 4x4 grid of 16x16 WMMA accumulators
// -> 16 WMMAs per 16 ds_load_b128 per K-step (2x the density of 32x64).
// MODE: 0 = bf16 row-major out, 1 = f32 row-major out,
//       2 = bf16 transposed out [b, col, s] (for V -> attention layout)
#define BM 256
#define BN 128
#define BK 32
#define ASTR 48   // padded LDS row stride (elements); 96B = 6x16B -> aligned
#define BSTR 48

template <int MODE>
__global__ __launch_bounds__(256) void k_gemm(const bf16_t* __restrict__ A,
                                              const bf16_t* __restrict__ BT,
                                              const float* __restrict__ bias,
                                              void* __restrict__ Cout,
                                              int M, int N, int K, float alpha) {
  __shared__ bf16_t As[BM * ASTR];   // 24 KB
  __shared__ bf16_t Bs[BN * BSTR];   // 12 KB  ([n][k], BT already transposed)

  const int tid  = threadIdx.x;
  const int lane = tid & 31;
  const int half = lane >> 4;
  const int l16  = lane & 15;
  const int wave = tid >> 5;
  const int waveM = wave >> 1;  // 0..3
  const int waveN = wave & 1;   // 0..1
  const int bm = blockIdx.y * BM;
  const int bn = blockIdx.x * BN;

  v8f acc[4][4];
#pragma unroll
  for (int tm = 0; tm < 4; ++tm)
#pragma unroll
    for (int tn = 0; tn < 4; ++tn) acc[tm][tn] = v8f{};

  for (int k0 = 0; k0 < K; k0 += BK) {
    // ---- stage A tile 256x32 (4 chunks) and B tile 128x32 (2 chunks) ----
#pragma unroll
    for (int c = 0; c < 4; ++c) {
      int e = (c * 256 + tid) * 8;          // 0..8184
      int r = e >> 5, kc = e & 31;
      const bf16_t* srcA = A + (size_t)(bm + r) * K + k0 + kc;
      *(v8bf*)&As[r * ASTR + kc] = *(const v8bf*)srcA;
      if (k0 + BK < K) __builtin_prefetch(srcA + BK, 0, 1);
    }
#pragma unroll
    for (int c = 0; c < 2; ++c) {
      int e = (c * 256 + tid) * 8;          // 0..4088
      int r = e >> 5, kc = e & 31;
      const bf16_t* srcB = BT + (size_t)(bn + r) * K + k0 + kc;
      *(v8bf*)&Bs[r * BSTR + kc] = *(const v8bf*)srcB;
      if (k0 + BK < K) __builtin_prefetch(srcB + BK, 0, 1);
    }
    __syncthreads();

    // ---- fragments ----
    v16bf afrag[4], bfrag[4];
#pragma unroll
    for (int tm = 0; tm < 4; ++tm) {
      int row = waveM * 64 + tm * 16 + l16;            // A: M = lane%16
      afrag[tm] = load_frag(&As[row * ASTR + half * 8],
                            &As[row * ASTR + 16 + half * 8]);
    }
#pragma unroll
    for (int tn = 0; tn < 4; ++tn) {
      int col = waveN * 64 + tn * 16 + l16;            // B: N = lane%16
      bfrag[tn] = load_frag(&Bs[col * BSTR + half * 8],
                            &Bs[col * BSTR + 16 + half * 8]);
    }
#pragma unroll
    for (int tm = 0; tm < 4; ++tm)
#pragma unroll
      for (int tn = 0; tn < 4; ++tn)
        acc[tm][tn] = wmma_bf16(afrag[tm], bfrag[tn], acc[tm][tn]);
    __syncthreads();
  }

  // ---- epilogue: C rows M = v + 8*half, cols N = lane%16 ----
#pragma unroll
  for (int tm = 0; tm < 4; ++tm)
#pragma unroll
    for (int tn = 0; tn < 4; ++tn)
#pragma unroll
      for (int v = 0; v < 8; ++v) {
        int row = bm + waveM * 64 + tm * 16 + v + half * 8;
        int col = bn + waveN * 64 + tn * 16 + l16;
        float val = (acc[tm][tn][v] + bias[col]) * alpha;
        if constexpr (MODE == 1) {
          ((float*)Cout)[(size_t)row * N + col] = val;
        } else if constexpr (MODE == 0) {
          ((bf16_t*)Cout)[(size_t)row * N + col] = f2bf(val);
        } else {  // MODE == 2: [b, col, s] with row = b*SS + s
          int b = row >> 11, s = row & (SS - 1);
          ((bf16_t*)Cout)[((size_t)b * KVDIM + col) * SS + s] = f2bf(val);
        }
      }
}

// ---------------- flash attention -----------------------------------------
// block = 8 waves = 8 consecutive 16-query tiles of one (batch, head);
// K/V tiles (32 keys) staged cooperatively in LDS and reused by all 8 waves.
// softmax scale folded into Q during its GEMM epilogue.
#define KSTR 72   // Ks row stride (elems): 144B, 16B-aligned, conflict-free
#define VSTR 40   // Vs row stride (elems):  80B, 16B-aligned, conflict-free

__global__ __launch_bounds__(256) void k_attn(const bf16_t* __restrict__ Q,
                                              const bf16_t* __restrict__ Kk,
                                              const bf16_t* __restrict__ Vt,
                                              bf16_t* __restrict__ O) {
  __shared__ bf16_t Ks[32 * KSTR];     // 4.5 KB, [key][d]
  __shared__ bf16_t Vs[64 * VSTR];     // 5.0 KB, [d][key]
  __shared__ bf16_t Plds[8][16 * 32];  // 8.0 KB, per-wave P transpose scratch

  const int tid  = threadIdx.x;
  const int lane = tid & 31;
  const int half = lane >> 4;
  const int l16  = lane & 15;
  const int wave = tid >> 5;

  const int blk = blockIdx.x;          // BB*NHEADS*(SS/128) = 1024 blocks
  const int qb  = blk & 15;
  const int h   = (blk >> 4) & (NHEADS - 1);
  const int b   = blk >> 9;
  const int kvh = h >> 2;
  const int q0  = qb * 128 + wave * 16;

  // Q A-fragments (16x64 split into two 16x32 k-slices); scale pre-folded.
  v16bf qf[2];
  const bf16_t* qbase = Q + ((size_t)(b * SS + q0 + l16)) * HIDDEN + h * HD;
#pragma unroll
  for (int kd = 0; kd < 2; ++kd)
    qf[kd] = load_frag(qbase + kd * 32 + half * 8,
                       qbase + kd * 32 + 16 + half * 8);

  v8f acc[4];
#pragma unroll
  for (int tt = 0; tt < 4; ++tt) acc[tt] = v8f{};
  float mrow[8], lrow[8];
#pragma unroll
  for (int v = 0; v < 8; ++v) { mrow[v] = -1e30f; lrow[v] = 0.0f; }

  const bf16_t* kbase = Kk + (size_t)(b * SS) * KVDIM + kvh * HD;
  const bf16_t* vbase = Vt + ((size_t)b * KVDIM + kvh * HD) * SS;
  bf16_t* P = &Plds[wave][0];

  // staging indices (one 16B chunk each for K and V per thread)
  const int kRow = (tid * 8) >> 6, kCol = (tid * 8) & 63;   // Ks: 32x64
  const int vRow = (tid * 8) >> 5, vCol = (tid * 8) & 31;   // Vs: 64x32

  for (int kb = 0; kb < SS; kb += 32) {
    // ---- cooperative staging of K (32x64) and V (64x32) tiles ----
    {
      const bf16_t* srcK = kbase + (size_t)(kb + kRow) * KVDIM + kCol;
      const bf16_t* srcV = vbase + (size_t)vRow * SS + kb + vCol;
      *(v8bf*)&Ks[kRow * KSTR + kCol] = *(const v8bf*)srcK;
      *(v8bf*)&Vs[vRow * VSTR + vCol] = *(const v8bf*)srcV;
      if (kb + 32 < SS) {
        __builtin_prefetch(srcK + 32 * KVDIM, 0, 1);
        __builtin_prefetch(srcV + 32, 0, 1);
      }
    }
    __syncthreads();

    // ---- scores: two 16(q)x16(k) tiles, K-dim 64 -> 2 WMMAs each ----
    v8f sc[2] = {v8f{}, v8f{}};
#pragma unroll
    for (int n = 0; n < 2; ++n) {
      const bf16_t* kr = &Ks[(n * 16 + l16) * KSTR];
#pragma unroll
      for (int kd = 0; kd < 2; ++kd) {
        v16bf bf = load_frag(kr + kd * 32 + half * 8,
                             kr + kd * 32 + 16 + half * 8);   // B = K^T
        sc[n] = wmma_bf16(qf[kd], bf, sc[n]);
      }
    }

    // ---- online softmax over the 32 new keys (DPP reductions) ----
    float ascale[8];
#pragma unroll
    for (int v = 0; v < 8; ++v) {
      float x = red_max16(fmaxf(sc[0][v], sc[1][v]));
      float mnew = fmaxf(mrow[v], x);
      float a  = __expf(mrow[v] - mnew);
      float p0 = __expf(sc[0][v] - mnew);
      float p1 = __expf(sc[1][v] - mnew);
      float rs = red_sum16(p0 + p1);
      lrow[v] = lrow[v] * a + rs;
      mrow[v] = mnew;
      ascale[v] = a;
      // spill P (C-layout: row = v+8*half, col = lane%16) to LDS as bf16
      int prow = v + half * 8;
      P[prow * 32 + l16]      = f2bf(p0);
      P[prow * 32 + 16 + l16] = f2bf(p1);
    }
#pragma unroll
    for (int tt = 0; tt < 4; ++tt)
#pragma unroll
      for (int v = 0; v < 8; ++v) acc[tt][v] *= ascale[v];

    // per-wave LDS ops complete in order; wait + compiler fence before
    // re-reading P in A-fragment layout.
    __builtin_amdgcn_wave_barrier();
    asm volatile("s_wait_dscnt 0" ::: "memory");

    v16bf pf = load_frag(P + l16 * 32 + half * 8,
                         P + l16 * 32 + 16 + half * 8);

    // ---- O += P(16x32) x V(32x64) ----
#pragma unroll
    for (int tt = 0; tt < 4; ++tt) {
      const bf16_t* vr = &Vs[(tt * 16 + l16) * VSTR];
      v16bf vf = load_frag(vr + half * 8, vr + 16 + half * 8);
      acc[tt] = wmma_bf16(pf, vf, acc[tt]);
    }
    __syncthreads();
  }

  // ---- normalize and store (bf16, [b,s,h,d]) ----
#pragma unroll
  for (int v = 0; v < 8; ++v) {
    float inv = 1.0f / lrow[v];
    int row = q0 + v + half * 8;
    bf16_t* orow = O + ((size_t)(b * SS + row)) * HIDDEN + h * HD;
#pragma unroll
    for (int tt = 0; tt < 4; ++tt)
      orow[tt * 16 + l16] = f2bf(acc[tt][v] * inv);
  }
}

// ---------------- launcher -------------------------------------------------
extern "C" void kernel_launch(void* const* d_in, const int* in_sizes, int n_in,
                              void* d_out, int out_size, void* d_ws, size_t ws_size,
                              hipStream_t stream) {
  (void)in_sizes; (void)n_in; (void)out_size; (void)ws_size;
  const float* x  = (const float*)d_in[0];
  const float* Wq = (const float*)d_in[1];
  const float* bq = (const float*)d_in[2];
  const float* Wk = (const float*)d_in[3];
  const float* bk = (const float*)d_in[4];
  const float* Wv = (const float*)d_in[5];
  const float* bv = (const float*)d_in[6];
  const float* Wo = (const float*)d_in[7];
  const float* bo = (const float*)d_in[8];
  float* out = (float*)d_out;

  char* ws = (char*)d_ws;
  size_t off = 0;
  auto alloc = [&](size_t bytes) -> void* {
    void* p = ws + off;
    off += (bytes + 255) & ~(size_t)255;
    return p;
  };
  bf16_t* xb  = (bf16_t*)alloc((size_t)MTOT * HIDDEN * 2);
  bf16_t* WqT = (bf16_t*)alloc((size_t)HIDDEN * HIDDEN * 2);   // [N][K]
  bf16_t* WkT = (bf16_t*)alloc((size_t)HIDDEN * KVDIM * 2);
  bf16_t* WvT = (bf16_t*)alloc((size_t)HIDDEN * KVDIM * 2);
  bf16_t* WoT = (bf16_t*)alloc((size_t)HIDDEN * HIDDEN * 2);
  bf16_t* Qb  = (bf16_t*)alloc((size_t)MTOT * HIDDEN * 2);
  bf16_t* Kb  = (bf16_t*)alloc((size_t)MTOT * KVDIM * 2);
  bf16_t* Vtb = (bf16_t*)alloc((size_t)MTOT * KVDIM * 2);      // [b][c][s]
  bf16_t* Ob  = (bf16_t*)alloc((size_t)MTOT * HIDDEN * 2);

  {  // x: plain row-major convert
    int n4 = (int)((size_t)MTOT * HIDDEN / 4);
    k_cvt<<<(n4 + 255) / 256, 256, 0, stream>>>(x, xb, n4);
  }
  auto cvtT = [&](const float* src, bf16_t* dst, int K, int N) {
    int n = K * N;
    k_cvt_t<<<(n + 255) / 256, 256, 0, stream>>>(src, dst, K, N);
  };
  cvtT(Wq, WqT, HIDDEN, HIDDEN);
  cvtT(Wk, WkT, HIDDEN, KVDIM);
  cvtT(Wv, WvT, HIDDEN, KVDIM);
  cvtT(Wo, WoT, HIDDEN, HIDDEN);

  // Q = (x@Wq + bq) * (1/sqrt(64)); K = x@Wk + bk; V = x@Wv + bv (-> [b,c,s])
  k_gemm<0><<<dim3(HIDDEN / BN, MTOT / BM), 256, 0, stream>>>(
      xb, WqT, bq, Qb, MTOT, HIDDEN, HIDDEN, 0.125f);
  k_gemm<0><<<dim3(KVDIM / BN, MTOT / BM), 256, 0, stream>>>(
      xb, WkT, bk, Kb, MTOT, KVDIM, HIDDEN, 1.0f);
  k_gemm<2><<<dim3(KVDIM / BN, MTOT / BM), 256, 0, stream>>>(
      xb, WvT, bv, Vtb, MTOT, KVDIM, HIDDEN, 1.0f);

  k_attn<<<BB * NHEADS * (SS / 128), 256, 0, stream>>>(Qb, Kb, Vtb, Ob);

  // out = attn @ Wo + bo (f32)
  k_gemm<1><<<dim3(HIDDEN / BN, MTOT / BM), 256, 0, stream>>>(
      Ob, WoT, bo, out, MTOT, HIDDEN, HIDDEN, 1.0f);
}
